// ProportionalAttentionWrapper_52278341927305
// MI455X (gfx1250) — matrix operand order repaired and checked by reading, not compiled, not run on hardware
//
#include <hip/hip_runtime.h>
#include <hip/hip_bf16.h>

#define B_ 2
#define S_ 4096
#define C_ 1024
#define H_ 16
#define DH_ 64
#define SCALE_F 2.8284271247461903f   // 64^0.25, folded into Q

typedef __attribute__((ext_vector_type(16))) _Float16 v16h;
typedef __attribute__((ext_vector_type(8)))  _Float16 v8h;
typedef __attribute__((ext_vector_type(8)))  float    v8f;
typedef __attribute__((ext_vector_type(4)))  float    v4f;

// Build a 16-half WMMA fragment from two contiguous 8-half (16B) chunks.
__device__ __forceinline__ v16h load_frag(const _Float16* lo, const _Float16* hi) {
  v8h a = *(const v8h*)lo;
  v8h b = *(const v8h*)hi;
  v16h f;
#pragma unroll
  for (int i = 0; i < 8; ++i) { f[i] = a[i]; f[i + 8] = b[i]; }
  return f;
}

// CDNA5 async DMA: copy 64 contiguous bytes (4 x b128, 16B per lane) from
// global to LDS, tracked by ASYNCcnt (no VGPR data round trip).
__device__ __forceinline__ void async_copy64(_Float16* ldst, const _Float16* gsrc) {
  unsigned l = (unsigned)(size_t)ldst;      // low 32 bits of generic = LDS addr
  const void* g = (const void*)gsrc;
  asm volatile("global_load_async_to_lds_b128 %0, %1, off"
               :: "v"(l), "v"(g) : "memory");
  asm volatile("global_load_async_to_lds_b128 %0, %1, off offset:16"
               :: "v"(l), "v"(g) : "memory");
  asm volatile("global_load_async_to_lds_b128 %0, %1, off offset:32"
               :: "v"(l), "v"(g) : "memory");
  asm volatile("global_load_async_to_lds_b128 %0, %1, off offset:48"
               :: "v"(l), "v"(g) : "memory");
}

__global__ void log_sizes_kernel(const int* __restrict__ ts,
                                 float* __restrict__ ls, int n) {
  int i = blockIdx.x * blockDim.x + threadIdx.x;
  if (i < n) ls[i] = logf((float)ts[i]);
}

// C = X(f32) @ W(f32) -> f16.
//   VT=false: head-split [B,H,S,DH] (Q, K), scaled.
//   VT=true : head-split transposed [B,H,DH,S] (V) -- packed b128 stores.
// Block: 128 thr (4 waves, 2x2 of 32x32), tile 64x64, K-step 32.
template <bool VT>
__global__ __launch_bounds__(128) void qkv_gemm_kernel(
    const float* __restrict__ X, const float* __restrict__ W,
    _Float16* __restrict__ out, float scale) {
  __shared__ _Float16 As[64 * 40];  // [m][k], padded stride 40 halfs
  __shared__ _Float16 Bs[64 * 40];  // [n][k] (W transposed), stride 40
  const int tid  = threadIdx.x;
  const int lane = tid & 31;
  const int wave = tid >> 5;
  const int l16  = lane & 15;
  const int hiL  = lane >> 4;
  const int kbA  = hiL * 8;   // A frag: K = {kbA..kbA+7, kbA+16..kbA+23}
  const int kbB  = hiL * 8;   // B frag: K = {kbB..kbB+7, kbB+16..kbB+23}
  const int wm   = (wave >> 1) * 32;
  const int wn   = (wave & 1) * 32;
  const int m0   = blockIdx.y * 64;
  const int n0   = blockIdx.x * 64;

  v8f acc[2][2] = {};

  for (int k0 = 0; k0 < C_; k0 += 32) {
    // Stage A 64x32: f32 -> f16, coalesced float4 loads.
#pragma unroll
    for (int p = 0; p < 4; ++p) {
      int r = p * 16 + (tid >> 3);
      int c = (tid & 7) * 4;
      v4f x = *(const v4f*)(X + (size_t)(m0 + r) * C_ + k0 + c);
      union { _Float16 h[4]; uint2 u; } t;
      t.h[0] = (_Float16)x[0]; t.h[1] = (_Float16)x[1];
      t.h[2] = (_Float16)x[2]; t.h[3] = (_Float16)x[3];
      *(uint2*)&As[r * 40 + c] = t.u;
    }
    // Stage W 32x64 transposed into Bs[n][k].
#pragma unroll
    for (int p = 0; p < 8; ++p) {
      int k = p * 4 + wave;
      int n = (tid & 31) * 2;
      const float* wp = W + (size_t)(k0 + k) * C_ + n0 + n;
      Bs[n * 40 + k]       = (_Float16)wp[0];
      Bs[(n + 1) * 40 + k] = (_Float16)wp[1];
    }
    __syncthreads();

    v16h af[2], bf[2];
#pragma unroll
    for (int i = 0; i < 2; ++i) {
      const _Float16* p = &As[(wm + i * 16 + l16) * 40 + kbA];
      af[i] = load_frag(p, p + 16);
    }
#pragma unroll
    for (int j = 0; j < 2; ++j) {
      const _Float16* p = &Bs[(wn + j * 16 + l16) * 40 + kbB];
      bf[j] = load_frag(p, p + 16);
    }
#pragma unroll
    for (int i = 0; i < 2; ++i)
#pragma unroll
      for (int j = 0; j < 2; ++j)
        acc[i][j] = __builtin_amdgcn_wmma_f32_16x16x32_f16(
            false, af[i], false, bf[j], (short)0, acc[i][j], false, false);
    __syncthreads();
  }

  // D layout: vgpr e, lane -> row = e + 8*(lane>=16), col = lane%16.
#pragma unroll
  for (int i = 0; i < 2; ++i)
#pragma unroll
    for (int j = 0; j < 2; ++j) {
      const int col = n0 + wn + j * 16 + l16;
      const int h   = col >> 6;
      const int dh  = col & 63;
      if (VT) {
        // Rows e are consecutive sequence positions -> one packed b128 store.
        const int base = m0 + wm + i * 16 + hiL * 8;
        const int b    = base >> 12;
        const int s    = base & (S_ - 1);
        v8h pk;
#pragma unroll
        for (int e = 0; e < 8; ++e) pk[e] = (_Float16)(acc[i][j][e] * scale);
        *(v8h*)(out + ((size_t)((b * H_ + h) * DH_ + dh)) * S_ + s) = pk;
      } else {
#pragma unroll
        for (int e = 0; e < 8; ++e) {
          const int row = m0 + wm + i * 16 + e + hiL * 8;
          const int b   = row >> 12;
          const int s   = row & (S_ - 1);
          out[((size_t)((b * H_ + h) * S_ + s) << 6) + dh] =
              (_Float16)(acc[i][j][e] * scale);
        }
      }
    }
}

// Flash attention: one block = (b, h, 64 query rows); 4 waves x 16 rows each.
// K ([key][dh]) and V^T ([dh][key], pre-transposed in global) tiles are
// double-buffered in LDS via GLOBAL_LOAD_ASYNC_TO_LDS_B128 (ASYNCcnt).
__global__ __launch_bounds__(128) void flash_attn_kernel(
    const _Float16* __restrict__ Qg, const _Float16* __restrict__ Kg,
    const _Float16* __restrict__ Vg, const float* __restrict__ Ls,
    _Float16* __restrict__ Og) {
  __shared__ _Float16 Kt[2][64 * 72];  // [key][dh], stride 72
  __shared__ _Float16 Vt[2][64 * 72];  // [dh][key], stride 72
  __shared__ _Float16 Ps[64 * 72];     // P tile [qrow][key], per-wave rows
  __shared__ float    lss[2][64];

  const int tid  = threadIdx.x;
  const int lane = tid & 31;
  const int wave = tid >> 5;
  const int l16  = lane & 15;
  const int hiL  = lane >> 4;
  const int kbA  = hiL * 8;
  const int kbB  = hiL * 8;

  const int b  = blockIdx.z;
  const int h  = blockIdx.y;
  const int q0 = blockIdx.x * 64;

  const _Float16* Qb = Qg + (((size_t)(b * H_ + h) * S_) << 6);
  const _Float16* Kb = Kg + (((size_t)(b * H_ + h) * S_) << 6);
  const _Float16* Vb = Vg + ((size_t)(b * H_ + h) * DH_) * S_;  // [dh][s]

  // Per-thread staging slice: 64B of one row (row = key for K, dh for V^T).
  const int srow = tid >> 1;
  const int scol = (tid & 1) * 32;

  // Q fragments (A-matrix) for this wave's 16 query rows, K-dim = DH = 2x32.
  v16h qf[2];
#pragma unroll
  for (int ks = 0; ks < 2; ++ks) {
    const _Float16* p = Qb + (size_t)(q0 + wave * 16 + l16) * DH_ + ks * 32 + kbA;
    qf[ks] = load_frag(p, p + 16);
  }

  v8f oacc[4] = {};
  float run_m[8], run_l[8];
#pragma unroll
  for (int j = 0; j < 8; ++j) { run_m[j] = -3.0e38f; run_l[j] = 0.0f; }

  // Prologue: async-stage tile 0 into buffer 0.
  async_copy64(&Kt[0][srow * 72 + scol], Kb + (size_t)srow * DH_ + scol);
  async_copy64(&Vt[0][srow * 72 + scol], Vb + (size_t)srow * S_ + scol);
  if (tid < 64) lss[0][tid] = Ls[b * S_ + tid];

  const int NT = S_ / 64;
  for (int it = 0; it < NT; ++it) {
    const int cur = it & 1;
    const int kt  = it * 64;
    // Own async fills done, then all waves' fills visible / prev reads done.
    asm volatile("s_wait_asynccnt 0" ::: "memory");
    __syncthreads();

    // Kick off next tile into the other buffer (overlaps with WMMA below).
    if (it + 1 < NT) {
      const int nb = cur ^ 1;
      async_copy64(&Kt[nb][srow * 72 + scol],
                   Kb + (size_t)(kt + 64 + srow) * DH_ + scol);
      async_copy64(&Vt[nb][srow * 72 + scol],
                   Vb + (size_t)srow * S_ + kt + 64 + scol);
      if (tid < 64) lss[nb][tid] = Ls[b * S_ + kt + 64 + tid];
    }

    // ---- scores = Q @ K^T  (4 key subtiles x 2 dh chunks) ----
    v8f sacc[4] = {};
#pragma unroll
    for (int nst = 0; nst < 4; ++nst)
#pragma unroll
      for (int ks = 0; ks < 2; ++ks) {
        const _Float16* p = &Kt[cur][(nst * 16 + l16) * 72 + ks * 32 + kbB];
        v16h kfrag = load_frag(p, p + 16);
        sacc[nst] = __builtin_amdgcn_wmma_f32_16x16x32_f16(
            false, qf[ks], false, kfrag, (short)0, sacc[nst], false, false);
      }

    // ---- + log(token_size) bias, online softmax in D layout ----
    float pm[8];
#pragma unroll
    for (int j = 0; j < 8; ++j) pm[j] = -3.0e38f;
#pragma unroll
    for (int nst = 0; nst < 4; ++nst) {
      const float bias = lss[cur][nst * 16 + l16];
#pragma unroll
      for (int j = 0; j < 8; ++j) {
        float s = sacc[nst][j] + bias;
        sacc[nst][j] = s;
        pm[j] = fmaxf(pm[j], s);
      }
    }
#pragma unroll
    for (int j = 0; j < 8; ++j)
#pragma unroll
      for (int m = 1; m < 16; m <<= 1)
        pm[j] = fmaxf(pm[j], __shfl_xor(pm[j], m, 32));

    float alpha[8], rsum[8];
#pragma unroll
    for (int j = 0; j < 8; ++j) {
      float nm = fmaxf(run_m[j], pm[j]);
      alpha[j] = __expf(run_m[j] - nm);
      run_m[j] = nm;
      rsum[j]  = 0.0f;
    }
#pragma unroll
    for (int nst = 0; nst < 4; ++nst)
#pragma unroll
      for (int j = 0; j < 8; ++j) {
        float p = __expf(sacc[nst][j] - run_m[j]);
        sacc[nst][j] = p;
        rsum[j] += p;
      }
#pragma unroll
    for (int j = 0; j < 8; ++j) {
#pragma unroll
      for (int m = 1; m < 16; m <<= 1) rsum[j] += __shfl_xor(rsum[j], m, 32);
      run_l[j] = run_l[j] * alpha[j] + rsum[j];
    }
#pragma unroll
    for (int t = 0; t < 4; ++t)
#pragma unroll
      for (int j = 0; j < 8; ++j) oacc[t][j] *= alpha[j];

    // ---- P: D layout -> row-major LDS (per-wave private rows) ----
#pragma unroll
    for (int nst = 0; nst < 4; ++nst)
#pragma unroll
      for (int j = 0; j < 8; ++j)
        Ps[(wave * 16 + j + 8 * hiL) * 72 + nst * 16 + l16] =
            (_Float16)sacc[nst][j];
    // Same-wave LDS RAW across lanes: drain DS counter (CDNA5 split waits).
    asm volatile("s_wait_dscnt 0" ::: "memory");

    // ---- O += P @ V ----
#pragma unroll
    for (int nst = 0; nst < 4; ++nst)
#pragma unroll
      for (int ks = 0; ks < 2; ++ks) {
        const _Float16* pp = &Ps[(wave * 16 + l16) * 72 + ks * 32 + kbA];
        v16h pfrag = load_frag(pp, pp + 16);
        const _Float16* vp = &Vt[cur][(nst * 16 + l16) * 72 + ks * 32 + kbB];
        v16h vfrag = load_frag(vp, vp + 16);
        oacc[nst] = __builtin_amdgcn_wmma_f32_16x16x32_f16(
            false, pfrag, false, vfrag, (short)0, oacc[nst], false, false);
      }
  }

  float inv[8];
#pragma unroll
  for (int j = 0; j < 8; ++j) inv[j] = 1.0f / run_l[j];
#pragma unroll
  for (int nst = 0; nst < 4; ++nst)
#pragma unroll
    for (int j = 0; j < 8; ++j) {
      const int row = q0 + wave * 16 + j + 8 * hiL;
      const int dh  = nst * 16 + l16;
      Og[(((size_t)(b * S_ + row)) << 10) + h * DH_ + dh] =
          (_Float16)(oacc[nst][j] * inv[j]);
    }
}

// out = O(f16, merged [B*S, C]) @ Wo(f32) + bo, f32 result.
__global__ __launch_bounds__(128) void out_proj_kernel(
    const _Float16* __restrict__ A16, const float* __restrict__ W,
    const float* __restrict__ bias, float* __restrict__ out) {
  __shared__ _Float16 As[64 * 40];
  __shared__ _Float16 Bs[64 * 40];
  const int tid  = threadIdx.x;
  const int lane = tid & 31;
  const int wave = tid >> 5;
  const int l16  = lane & 15;
  const int hiL  = lane >> 4;
  const int kbA  = hiL * 8;
  const int kbB  = hiL * 8;
  const int wm   = (wave >> 1) * 32;
  const int wn   = (wave & 1) * 32;
  const int m0   = blockIdx.y * 64;
  const int n0   = blockIdx.x * 64;

  v8f acc[2][2] = {};

  for (int k0 = 0; k0 < C_; k0 += 32) {
#pragma unroll
    for (int p = 0; p < 4; ++p) {
      int r = p * 16 + (tid >> 3);
      int c = (tid & 7) * 4;
      *(uint2*)&As[r * 40 + c] =
          *(const uint2*)(A16 + (size_t)(m0 + r) * C_ + k0 + c);
    }
#pragma unroll
    for (int p = 0; p < 8; ++p) {
      int k = p * 4 + wave;
      int n = (tid & 31) * 2;
      const float* wp = W + (size_t)(k0 + k) * C_ + n0 + n;
      Bs[n * 40 + k]       = (_Float16)wp[0];
      Bs[(n + 1) * 40 + k] = (_Float16)wp[1];
    }
    __syncthreads();

    v16h af[2], bf[2];
#pragma unroll
    for (int i = 0; i < 2; ++i) {
      const _Float16* p = &As[(wm + i * 16 + l16) * 40 + kbA];
      af[i] = load_frag(p, p + 16);
    }
#pragma unroll
    for (int j = 0; j < 2; ++j) {
      const _Float16* p = &Bs[(wn + j * 16 + l16) * 40 + kbB];
      bf[j] = load_frag(p, p + 16);
    }
#pragma unroll
    for (int i = 0; i < 2; ++i)
#pragma unroll
      for (int j = 0; j < 2; ++j)
        acc[i][j] = __builtin_amdgcn_wmma_f32_16x16x32_f16(
            false, af[i], false, bf[j], (short)0, acc[i][j], false, false);
    __syncthreads();
  }

#pragma unroll
  for (int i = 0; i < 2; ++i)
#pragma unroll
    for (int j = 0; j < 2; ++j) {
      const int col = n0 + wn + j * 16 + l16;
      const float bv = bias[col];
#pragma unroll
      for (int e = 0; e < 8; ++e) {
        const int row = m0 + wm + i * 16 + e + hiL * 8;
        out[(size_t)row * C_ + col] = acc[i][j][e] + bv;
      }
    }
}

extern "C" void kernel_launch(void* const* d_in, const int* in_sizes, int n_in,
                              void* d_out, int out_size, void* d_ws,
                              size_t ws_size, hipStream_t stream) {
  const float* X  = (const float*)d_in[0];
  const float* Wq = (const float*)d_in[1];
  const float* Wk = (const float*)d_in[2];
  const float* Wv = (const float*)d_in[3];
  const float* Wo = (const float*)d_in[4];
  const float* bo = (const float*)d_in[5];
  const int*   ts = (const int*)d_in[6];

  const size_t NE = (size_t)B_ * S_ * C_;  // elements per activation tensor
  _Float16* Qf = (_Float16*)d_ws;
  _Float16* Kf = Qf + NE;
  _Float16* Vf = Kf + NE;   // stored [B,H,DH,S]
  _Float16* Of = Vf + NE;
  float*    Ls = (float*)(Of + NE);

  log_sizes_kernel<<<dim3((B_ * S_ + 255) / 256), dim3(256), 0, stream>>>(
      ts, Ls, B_ * S_);

  dim3 gg(C_ / 64, (B_ * S_) / 64), bb(128);
  qkv_gemm_kernel<false><<<gg, bb, 0, stream>>>(X, Wq, Qf, SCALE_F);
  qkv_gemm_kernel<false><<<gg, bb, 0, stream>>>(X, Wk, Kf, 1.0f);
  qkv_gemm_kernel<true ><<<gg, bb, 0, stream>>>(X, Wv, Vf, 1.0f);

  flash_attn_kernel<<<dim3(S_ / 64, H_, B_), dim3(128), 0, stream>>>(
      Qf, Kf, Vf, Ls, Of);

  out_proj_kernel<<<gg, bb, 0, stream>>>(Of, Wo, bo, (float*)d_out);
}
